// MultiHeadAttention_20572893348027
// MI455X (gfx1250) — matrix operand order repaired and checked by reading, not compiled, not run on hardware
//
#include <hip/hip_runtime.h>

#define DEV __device__ __forceinline__

typedef __bf16 bf16_t;
typedef bf16_t v16bf __attribute__((ext_vector_type(16)));
typedef float  v8f   __attribute__((ext_vector_type(8)));
typedef unsigned int v4u __attribute__((ext_vector_type(4)));
typedef int v4i __attribute__((ext_vector_type(4)));
typedef int v8i __attribute__((ext_vector_type(8)));

struct __align__(16) U128 { unsigned int x, y, z, w; };

union FragBF {
  U128 u[2];
  v16bf v;
  unsigned short s[16];
};

DEV U128 ld128(const unsigned short* p) {
  return *reinterpret_cast<const U128*>(p);
}

DEV unsigned short f2bf(float f) {
  unsigned int u = __float_as_uint(f);
  u += 0x7FFFu + ((u >> 16) & 1u);           // round-to-nearest-even
  return (unsigned short)(u >> 16);
}

DEV v8f wmma_bf16(const FragBF& a, const FragBF& b, v8f c) {
  return __builtin_amdgcn_wmma_f32_16x16x32_bf16(false, a.v, false, b.v,
                                                 (short)0, c, false, false);
}

// ---- 16-lane-half reductions: v_permlane16_b32 (VALU) with shfl fallback ----
#if __has_builtin(__builtin_amdgcn_permlane16)
DEV float pl16(float x, unsigned lo, unsigned hi) {
  return __uint_as_float(__builtin_amdgcn_permlane16(
      __float_as_uint(x), __float_as_uint(x), lo, hi, false, false));
}
DEV float red_max16(float x) {
  x = fmaxf(x, pl16(x, 0x67452301u, 0xEFCDAB89u));  // xor 1
  x = fmaxf(x, pl16(x, 0x54761032u, 0xDCFE98BAu));  // xor 2
  x = fmaxf(x, pl16(x, 0x32107654u, 0xBA98FEDCu));  // xor 4
  x = fmaxf(x, pl16(x, 0xFEDCBA98u, 0x76543210u));  // xor 8
  return x;
}
DEV float red_sum16(float x) {
  x += pl16(x, 0x67452301u, 0xEFCDAB89u);
  x += pl16(x, 0x54761032u, 0xDCFE98BAu);
  x += pl16(x, 0x32107654u, 0xBA98FEDCu);
  x += pl16(x, 0xFEDCBA98u, 0x76543210u);
  return x;
}
#else
DEV float red_max16(float x) {
  for (int off = 8; off >= 1; off >>= 1) x = fmaxf(x, __shfl_xor(x, off, 32));
  return x;
}
DEV float red_sum16(float x) {
  for (int off = 8; off >= 1; off >>= 1) x += __shfl_xor(x, off, 32);
  return x;
}
#endif

// ---- Tensor Data Mover (2-D tile -> LDS), toolchain-arity hedged ----------
#if __has_builtin(__builtin_amdgcn_tensor_load_to_lds)
#define HAVE_TDM 1
DEV void tdm_load_tile2d(const void* gptr, unsigned lds_off, int rows,
                         int k_elems) {
  unsigned long long ga = (unsigned long long)gptr;
  v4u g0;
  g0[0] = 1u;                                   // count=1 (valid descriptor)
  g0[1] = lds_off;                              // LDS byte address
  g0[2] = (unsigned)ga;                         // global_addr[31:0]
  g0[3] = (unsigned)(ga >> 32) | (2u << 30);    // global_addr[56:32] | type=2
  v8i g1;
  g1[0] = 0x00010000;                           // data_size=1 (2 bytes)
  g1[1] = (64 << 16);                           // tensor_dim0 = 64 (== tile)
  g1[2] = (rows & 0xFFFF) << 16;                // tensor_dim1[15:0]
  g1[3] = (64 << 16);                           // dim1 hi=0 | tile_dim0 = 64
  g1[4] = rows;                                 // tile_dim1 | tile_dim2=0
  g1[5] = k_elems;                              // tensor_dim0_stride[31:0]
  g1[6] = 0;
  g1[7] = 0;
  v4i z4 = {0, 0, 0, 0};
#if __has_include(<hip/amd_detail/amd_gfx1250_TDM.h>)
  v8i z8 = {0, 0, 0, 0, 0, 0, 0, 0};
  __builtin_amdgcn_tensor_load_to_lds(g0, g1, z4, z4, z8, 0);  // clang-23 form
#else
  __builtin_amdgcn_tensor_load_to_lds(g0, g1, z4, z4, 0);      // ROCm 7.2 form
#endif
}
#else
#define HAVE_TDM 0
#endif

// ---------------------------------------------------------------------------
// f32 -> bf16 elementwise conversion
// ---------------------------------------------------------------------------
__global__ void __launch_bounds__(256)
f32_to_bf16_kernel(const float* __restrict__ in, unsigned short* __restrict__ out,
                   int n) {
  int i = blockIdx.x * 256 + threadIdx.x;
  if (i < n) out[i] = f2bf(in[i]);
}

// ---------------------------------------------------------------------------
// bf16 GEMM: C[m,n] = sum_k A[m][k]*B[n][k] + bias[n]
// Block 256 thr (8 waves), block tile 128x128, wave tile 64x32, K-chunk 64.
// LDS tiles double-buffered and filled by the Tensor Data Mover.
// MODE 0: f32 out.  MODE 1: QKV bf16 out (+0.125 q-scale, V^T side store).
// ---------------------------------------------------------------------------
template <int MODE>
__global__ void __launch_bounds__(256)
gemm_bf16_kernel(const unsigned short* __restrict__ A,
                 const unsigned short* __restrict__ Bm,
                 const float* __restrict__ bias,
                 void* __restrict__ outp,
                 unsigned short* __restrict__ vT,
                 int M, int N, int K)
{
  __shared__ __align__(16) unsigned char sA[2][128 * 128];  // 128 rows x 64 bf16
  __shared__ __align__(16) unsigned char sB[2][128 * 128];

  const int tid  = threadIdx.x;
  const int lane = tid & 31;
  const int wave = tid >> 5;
  const int wm   = wave & 1;   // 2 waves in M
  const int wn   = wave >> 1;  // 4 waves in N
  const int n16  = lane & 15;
  const int hh   = lane >> 4;

  const int n0 = blockIdx.x * 128;
  const int m0 = blockIdx.y * 128;

  const v8f vzero = {0.f,0.f,0.f,0.f,0.f,0.f,0.f,0.f};
  v8f acc[4][2];
#pragma unroll
  for (int i = 0; i < 4; ++i)
#pragma unroll
    for (int j = 0; j < 2; ++j) acc[i][j] = vzero;

  int cur = 0;
#if HAVE_TDM
  if (wave == 0) {
    tdm_load_tile2d(A + (size_t)m0 * K, (unsigned)(size_t)&sA[0][0], 128, K);
    tdm_load_tile2d(Bm + (size_t)n0 * K, (unsigned)(size_t)&sB[0][0], 128, K);
  }
#endif

  for (int k0 = 0; k0 < K; k0 += 64) {
#if HAVE_TDM
    if (wave == 0) __builtin_amdgcn_s_wait_tensorcnt(0);
    __syncthreads();
    if (wave == 0 && k0 + 64 < K) {   // prefetch next chunk into other buffer
      tdm_load_tile2d(A + (size_t)m0 * K + k0 + 64,
                      (unsigned)(size_t)&sA[cur ^ 1][0], 128, K);
      tdm_load_tile2d(Bm + (size_t)n0 * K + k0 + 64,
                      (unsigned)(size_t)&sB[cur ^ 1][0], 128, K);
    }
#else
    __syncthreads();
#pragma unroll
    for (int p = 0; p < 4; ++p) {
      int i = tid + p * 256;           // 1024 x 16B per tile
      int row = i >> 3, seg = i & 7;
      reinterpret_cast<U128*>(sA[cur])[i] =
          ld128(A + (size_t)(m0 + row) * K + k0 + seg * 8);
      reinterpret_cast<U128*>(sB[cur])[i] =
          ld128(Bm + (size_t)(n0 + row) * K + k0 + seg * 8);
    }
    __syncthreads();
#endif

#pragma unroll
    for (int kc = 0; kc < 2; ++kc) {
      FragBF fa[4];
#pragma unroll
      for (int ms = 0; ms < 4; ++ms) {
        const unsigned char* pr =
            sA[cur] + (wm * 64 + ms * 16 + n16) * 128 + kc * 64;
        fa[ms].u[0] = *reinterpret_cast<const U128*>(pr + 16 * hh);
        fa[ms].u[1] = *reinterpret_cast<const U128*>(pr + 32 + 16 * hh);
      }
#pragma unroll
      for (int j = 0; j < 2; ++j) {
        FragBF fb;
        const unsigned char* pb =
            sB[cur] + (wn * 32 + j * 16 + n16) * 128 + kc * 64;
        fb.u[0] = *reinterpret_cast<const U128*>(pb + 16 * hh);
        fb.u[1] = *reinterpret_cast<const U128*>(pb + 32 + 16 * hh);
#pragma unroll
        for (int ms = 0; ms < 4; ++ms)
          acc[ms][j] = wmma_bf16(fa[ms], fb, acc[ms][j]);
      }
    }
    cur ^= 1;
  }

  // Epilogue. C layout: lane = col n16, rows r + 8*hh.
#pragma unroll
  for (int j = 0; j < 2; ++j) {
    const int n = n0 + wn * 32 + j * 16 + n16;
    const float bv = bias[n];
#pragma unroll
    for (int ms = 0; ms < 4; ++ms) {
      const int mbase = m0 + wm * 64 + ms * 16 + 8 * hh;
      if (MODE == 0) {
        float* out = (float*)outp;
#pragma unroll
        for (int r = 0; r < 8; ++r)
          out[(size_t)(mbase + r) * N + n] = acc[ms][j][r] + bv;
      } else {
        unsigned short* qkv = (unsigned short*)outp;
        unsigned short tmp[8];
#pragma unroll
        for (int r = 0; r < 8; ++r) {
          float v = acc[ms][j][r] + bv;
          if (n < 1024) v *= 0.125f;     // fold 1/sqrt(HD) into Q
          unsigned short bs = f2bf(v);
          tmp[r] = bs;
          qkv[(size_t)(mbase + r) * N + n] = bs;
        }
        if (n >= 2048) {                 // V^T side store: vT[b][h][dim][tok]
          const int d = n - 2048, head = d >> 6, dd = d & 63;
          const int bidx = mbase >> 11, t = mbase & 2047;
          union { unsigned short s[8]; U128 u; } pk;
#pragma unroll
          for (int r = 0; r < 8; ++r) pk.s[r] = tmp[r];
          *reinterpret_cast<U128*>(
              vT + ((size_t)(bidx * 16 + head) * 64 + dd) * 2048 + t) = pk.u;
        }
      }
    }
  }
}

// ---------------------------------------------------------------------------
// Flash attention: one wave per 16-row Q tile, 64-key chunks, online softmax.
// ---------------------------------------------------------------------------
__global__ void __launch_bounds__(256)
attn_kernel(const unsigned short* __restrict__ qkv,
            const unsigned short* __restrict__ vT,
            unsigned short* __restrict__ ctx)
{
  const int T = 2048, Dm = 1024, NQ = 3072, H = 16;
  __shared__ __align__(16) unsigned char sP[8][2048];  // per-wave 16x64 bf16

  const int lane = threadIdx.x & 31;
  const int wave = threadIdx.x >> 5;
  const int n16  = lane & 15;
  const int hh   = lane >> 4;
  const int bh = blockIdx.y;
  const int b = bh >> 4, h = bh & 15;
  const int t0 = blockIdx.x * 128 + wave * 16;

  unsigned short* myP = (unsigned short*)sP[wave];

  FragBF fq[2];
  {
    const size_t qb = (size_t)(b * T + t0 + n16) * NQ + h * 64;
#pragma unroll
    for (int kc = 0; kc < 2; ++kc) {
      fq[kc].u[0] = ld128(qkv + qb + kc * 32 + 8 * hh);
      fq[kc].u[1] = ld128(qkv + qb + kc * 32 + 16 + 8 * hh);
    }
  }

  const v8f vzero = {0.f,0.f,0.f,0.f,0.f,0.f,0.f,0.f};
  v8f ctxa[4] = {vzero, vzero, vzero, vzero};
  float mi[8], li[8];
#pragma unroll
  for (int r = 0; r < 8; ++r) { mi[r] = -1e30f; li[r] = 0.f; }

  for (int kt = 0; kt < T; kt += 64) {
    // S = Q K^T for 64 keys (4 x 16-key column tiles)
    v8f s[4] = {vzero, vzero, vzero, vzero};
#pragma unroll
    for (int kc = 0; kc < 2; ++kc) {
#pragma unroll
      for (int nt = 0; nt < 4; ++nt) {
        const size_t kb = (size_t)(b * T + kt + nt * 16 + n16) * NQ + Dm +
                          h * 64 + kc * 32 + 8 * hh;
        FragBF fk;
        fk.u[0] = ld128(qkv + kb);
        fk.u[1] = ld128(qkv + kb + 16);
        s[nt] = wmma_bf16(fq[kc], fk, s[nt]);
      }
    }

    // Online softmax (permlane16 reductions within 16-lane halves)
#pragma unroll
    for (int r = 0; r < 8; ++r) {
      float a0 = s[0][r], a1 = s[1][r], a2 = s[2][r], a3 = s[3][r];
      float mx = red_max16(fmaxf(fmaxf(a0, a1), fmaxf(a2, a3)));
      mx = fmaxf(mx, mi[r]);
      const float corr = __expf(mi[r] - mx);
      mi[r] = mx;
      const float p0 = __expf(a0 - mx);
      const float p1 = __expf(a1 - mx);
      const float p2 = __expf(a2 - mx);
      const float p3 = __expf(a3 - mx);
      const float ps = red_sum16((p0 + p1) + (p2 + p3));
      li[r] = li[r] * corr + ps;
#pragma unroll
      for (int j = 0; j < 4; ++j) ctxa[j][r] *= corr;
      unsigned short* pw = myP + (r + 8 * hh) * 64 + n16;
      pw[0]  = f2bf(p0);
      pw[16] = f2bf(p1);
      pw[32] = f2bf(p2);
      pw[48] = f2bf(p3);
    }

    // Wave-private LDS tile: DS ops are in-order per wave; drain DScnt.
    __builtin_amdgcn_wave_barrier();
    asm volatile("s_wait_dscnt 0" ::: "memory");

    // Reload P as A fragments (16x64 -> two 32-K chunks)
    FragBF fp[2];
#pragma unroll
    for (int kc = 0; kc < 2; ++kc) {
      const unsigned short* pr = myP + n16 * 64 + kc * 32;
      fp[kc].u[0] = *reinterpret_cast<const U128*>(pr + 8 * hh);
      fp[kc].u[1] = *reinterpret_cast<const U128*>(pr + 16 + 8 * hh);
    }

    // ctx += P V  (V^T fragments contiguous thanks to vT layout)
#pragma unroll
    for (int kc = 0; kc < 2; ++kc) {
#pragma unroll
      for (int j = 0; j < 4; ++j) {
        const size_t vb = ((size_t)(b * H + h) * 64 + j * 16 + n16) * T + kt +
                          kc * 32 + 8 * hh;
        FragBF fv;
        fv.u[0] = ld128(vT + vb);
        fv.u[1] = ld128(vT + vb + 16);
        ctxa[j] = wmma_bf16(fp[kc], fv, ctxa[j]);
      }
    }
    __builtin_amdgcn_wave_barrier();
  }

  // Normalize and store merged-head context (bf16)
#pragma unroll
  for (int j = 0; j < 4; ++j) {
#pragma unroll
    for (int r = 0; r < 8; ++r) {
      const float v = ctxa[j][r] / li[r];
      ctx[(size_t)(b * T + t0 + r + 8 * hh) * Dm + h * 64 + j * 16 + n16] =
          f2bf(v);
    }
  }
}

// ---------------------------------------------------------------------------
// Launch: convert -> QKV GEMM -> flash attention -> output GEMM
// ---------------------------------------------------------------------------
extern "C" void kernel_launch(void* const* d_in, const int* in_sizes, int n_in,
                              void* d_out, int out_size, void* d_ws, size_t ws_size,
                              hipStream_t stream) {
  (void)in_sizes; (void)n_in; (void)out_size; (void)ws_size;

  const float* x     = (const float*)d_in[0];
  const float* w_qkv = (const float*)d_in[1];
  const float* b_qkv = (const float*)d_in[2];
  const float* w_out = (const float*)d_in[3];
  const float* b_out = (const float*)d_in[4];
  float* out = (float*)d_out;

  char* ws = (char*)d_ws;
  unsigned short* xb    = (unsigned short*)(ws + ((size_t)0  << 20));  //  8 MiB
  unsigned short* wqkvb = (unsigned short*)(ws + ((size_t)8  << 20));  //  6 MiB
  unsigned short* woutb = (unsigned short*)(ws + ((size_t)14 << 20));  //  2 MiB
  unsigned short* qkvb  = (unsigned short*)(ws + ((size_t)16 << 20));  // 24 MiB
  unsigned short* vTb   = (unsigned short*)(ws + ((size_t)40 << 20));  //  8 MiB
  unsigned short* ctxb  = (unsigned short*)(ws + ((size_t)48 << 20));  //  8 MiB

  const int B = 2, T = 2048, D = 1024;
  const int M = B * T;

  f32_to_bf16_kernel<<<(M * D + 255) / 256, 256, 0, stream>>>(x, xb, M * D);
  f32_to_bf16_kernel<<<(3 * D * D + 255) / 256, 256, 0, stream>>>(w_qkv, wqkvb, 3 * D * D);
  f32_to_bf16_kernel<<<(D * D + 255) / 256, 256, 0, stream>>>(w_out, woutb, D * D);

  gemm_bf16_kernel<1><<<dim3(3 * D / 128, M / 128), 256, 0, stream>>>(
      xb, wqkvb, b_qkv, (void*)qkvb, vTb, M, 3 * D, D);

  attn_kernel<<<dim3(T / 128, B * 16), 256, 0, stream>>>(qkvb, vTb, ctxb);

  gemm_bf16_kernel<0><<<dim3(D / 128, M / 128), 256, 0, stream>>>(
      ctxb, woutb, b_out, (void*)out, nullptr, M, D, D);
}
